// Dynamics_22385369547237
// MI455X (gfx1250) — matrix operand order repaired and checked by reading, not compiled, not run on hardware
//
#include <hip/hip_runtime.h>

#define XDIM 32
#define UDIM 8
#define ADIM 24
#define HDIM 256
#define TSTEPS 128
#define BATCH 256
#define EPB 8                 // batch elements per workgroup
#define NBLOCKS (BATCH / EPB) // 32 persistent workgroups
#define MIN_VC 0.01f
#define SPAN_VC 0.99f
#define EPS_VC 1e-6f

#define NP1 1312 // packed prior heads: WA(1024) + WB(256) + Wnx(32)
#define NP2 800  // packed post heads:  WC(768) + Wna(24) + pad(8)

typedef __attribute__((ext_vector_type(16))) _Float16 v16h;
typedef __attribute__((ext_vector_type(8))) float v8f;

union H16 {
  v16h v;
  _Float16 h[16];
  float4 f4[2];
};

__device__ inline v8f wmma_f16(v16h a, v16h b, v8f c) {
  // D = A(16x32 f16) * B(32x16 f16) + C(16x16 f32)
  return __builtin_amdgcn_wmma_f32_16x16x32_f16(false, a, false, b, (short)0, c,
                                                false, false);
}

// ---------------------------------------------------------------------------
// Pre-pass: convert weights to f16 and pack head weights/biases.
// For D = X @ W^T, lane n of a WMMA B tile holds 16 contiguous halves of W
// row (n0+n) -> plain row-major f16 needs no reorder; A uses the same contiguous k order.
// ---------------------------------------------------------------------------
__global__ void pack_weights(
    const float* __restrict__ W1, const float* __restrict__ W2,
    const float* __restrict__ WA, const float* __restrict__ WB,
    const float* __restrict__ WC, const float* __restrict__ Wnx,
    const float* __restrict__ Wna, const float* __restrict__ bA,
    const float* __restrict__ bB, const float* __restrict__ bnx,
    const float* __restrict__ bC, const float* __restrict__ bna,
    _Float16* __restrict__ W1h, _Float16* __restrict__ W2h,
    _Float16* __restrict__ Wp1, _Float16* __restrict__ Wp2,
    float* __restrict__ bp1, float* __restrict__ bp2) {
  const int stride = gridDim.x * blockDim.x;
  const int t0 = blockIdx.x * blockDim.x + threadIdx.x;
  for (int i = t0; i < HDIM * XDIM; i += stride) W1h[i] = (_Float16)W1[i];
  for (int i = t0; i < HDIM * HDIM; i += stride) W2h[i] = (_Float16)W2[i];
  for (int i = t0; i < NP1 * HDIM; i += stride) {
    int n = i / HDIM, k = i % HDIM;
    float v = (n < 1024)   ? WA[n * HDIM + k]
              : (n < 1280) ? WB[(n - 1024) * HDIM + k]
                           : Wnx[(n - 1280) * HDIM + k];
    Wp1[i] = (_Float16)v;
  }
  for (int i = t0; i < NP2 * HDIM; i += stride) {
    int n = i / HDIM, k = i % HDIM;
    float v = (n < 768) ? WC[n * HDIM + k]
              : (n < 792) ? Wna[(n - 768) * HDIM + k] : 0.f;
    Wp2[i] = (_Float16)v;
  }
  for (int i = t0; i < NP1; i += stride)
    bp1[i] = (i < 1024) ? bA[i] : (i < 1280) ? bB[i - 1024] : bnx[i - 1280];
  for (int i = t0; i < NP2; i += stride)
    bp2[i] = (i < 768) ? bC[i] : (i < 792) ? bna[i - 768] : 0.f;
}

// ---------------------------------------------------------------------------
// Persistent Kalman-scan kernel: each block owns EPB batch elements for all T.
// ---------------------------------------------------------------------------
__global__ __launch_bounds__(256, 1) void kf_scan_kernel(
    const float* __restrict__ u, const float* __restrict__ a,
    const float* __restrict__ b1, const float* __restrict__ b2,
    const float* __restrict__ bp1, const float* __restrict__ bp2,
    const float* __restrict__ alpha, const _Float16* __restrict__ W1h,
    const _Float16* __restrict__ W2h, const _Float16* __restrict__ Wp1,
    const _Float16* __restrict__ Wp2, float* __restrict__ out) {
  __shared__ __align__(16) float s_qm[EPB][XDIM];
  __shared__ __align__(16) float s_prm[EPB][XDIM];
  __shared__ __align__(16) float s_qc[EPB][XDIM * XDIM];
  __shared__ __align__(16) float s_prc[EPB][XDIM * XDIM];
  __shared__ __align__(16) float s_bigA[EPB][XDIM * XDIM]; // Am / Cm
  __shared__ __align__(16) float s_bigT[EPB][XDIM * XDIM]; // Cm@prc (post)
  __shared__ __align__(16) float s_raw[EPB][XDIM * XDIM];
  __shared__ __align__(16) float s_S[EPB][ADIM * ADIM]; // S, inverted in place
  __shared__ __align__(16) _Float16 s_h0[EPB][HDIM];
  __shared__ __align__(16) _Float16 s_h1[EPB][HDIM];
  __shared__ float s_Bm[EPB][XDIM * UDIM];
  __shared__ float s_nx[EPB][XDIM];
  __shared__ float s_na[EPB][ADIM];
  __shared__ float s_iv[EPB][ADIM];
  __shared__ float s_b1[HDIM], s_b2[HDIM];
  __shared__ float s_bp1[NP1], s_bp2[NP2];

  const int tid = threadIdx.x;
  const int lane = tid & 31;
  const int wv = tid >> 5; // wave id; also the batch element for VALU phases
  const int e = wv;
  const int i = lane;
  const int bglob = blockIdx.x * EPB + e;
  const float alpha0 = alpha[0];

  const size_t OPM = 0;
  const size_t OPC = (size_t)TSTEPS * BATCH * XDIM;
  const size_t OQM = OPC + (size_t)TSTEPS * BATCH * XDIM * XDIM;
  const size_t OQC = OQM + (size_t)TSTEPS * BATCH * XDIM;

  // Stage biases in LDS once (reused 128x).
  for (int q = tid; q < HDIM; q += 256) {
    s_b1[q] = b1[q];
    s_b2[q] = b2[q];
  }
  for (int q = tid; q < NP1; q += 256) s_bp1[q] = bp1[q];
  for (int q = tid; q < NP2; q += 256) s_bp2[q] = bp2[q];

  // ---- dyn: x -> h (two WMMA GEMMs) -> heads (WMMA GEMM), phase 0=prior
  auto dyn = [&](const float* xm, int phase) {
    __syncthreads();
    // GEMM1: h0 = relu(x @ W1^T + b1), K = 32 (single k-tile)
    {
      H16 a0;
      const int m = lane & 15, koff = (lane >> 4) * 16;
      if (m < EPB) {
#pragma unroll
        for (int q = 0; q < 8; ++q) {
          a0.h[q] = (_Float16)xm[m * XDIM + koff + q];
          a0.h[q + 8] = (_Float16)xm[m * XDIM + koff + 8 + q];
        }
      } else {
#pragma unroll
        for (int q = 0; q < 16; ++q) a0.h[q] = (_Float16)0.f;
      }
      for (int nt = wv; nt < HDIM / 16; nt += 8) {
        const int n0 = nt * 16;
        H16 bt;
        const _Float16* wp =
            W1h + (size_t)(n0 + (lane & 15)) * XDIM + (lane >> 4) * 16;
        bt.f4[0] = ((const float4*)wp)[0];
        bt.f4[1] = ((const float4*)wp)[1];
        v8f c;
#pragma unroll
        for (int r = 0; r < 8; ++r) c[r] = 0.f;
        c = wmma_f16(a0.v, bt.v, c);
        if (lane < 16) {
          const int n = n0 + lane;
          const float bias = s_b1[n];
#pragma unroll
          for (int r = 0; r < 8; ++r) {
            float v = c[r] + bias;
            s_h0[r][n] = (_Float16)(v > 0.f ? v : 0.f);
          }
        }
      }
    }
    __syncthreads();
    // GEMM2: h1 = relu(h0 @ W2^T + b2), K = 256
    {
      H16 at[8];
      const int m = lane & 15, koff = (lane >> 4) * 16;
#pragma unroll
      for (int kt = 0; kt < 8; ++kt) {
        if (m < EPB) {
          const _Float16* hp = &s_h0[m][kt * 32 + koff];
          at[kt].f4[0] = ((const float4*)hp)[0];
          at[kt].f4[1] = ((const float4*)(hp + 8))[0];
        } else {
#pragma unroll
          for (int q = 0; q < 16; ++q) at[kt].h[q] = (_Float16)0.f;
        }
      }
      for (int nt = wv; nt < HDIM / 16; nt += 8) {
        const int n0 = nt * 16;
        const _Float16* wrow =
            W2h + (size_t)(n0 + (lane & 15)) * HDIM + (lane >> 4) * 16;
        H16 bt[8]; // hoist all B tiles: one big load clause, overlapped waits
#pragma unroll
        for (int kt = 0; kt < 8; ++kt) {
          bt[kt].f4[0] = ((const float4*)(wrow + kt * 32))[0];
          bt[kt].f4[1] = ((const float4*)(wrow + kt * 32 + 8))[0];
        }
        v8f c;
#pragma unroll
        for (int r = 0; r < 8; ++r) c[r] = 0.f;
#pragma unroll
        for (int kt = 0; kt < 8; ++kt) c = wmma_f16(at[kt].v, bt[kt].v, c);
        if (lane < 16) {
          const int n = n0 + lane;
          const float bias = s_b2[n];
#pragma unroll
          for (int r = 0; r < 8; ++r) {
            float v = c[r] + bias;
            s_h1[r][n] = (_Float16)(v > 0.f ? v : 0.f);
          }
        }
      }
    }
    __syncthreads();
    // Heads: phase-specific packed weight matrix, K = 256
    {
      H16 at[8];
      const int m = lane & 15, koff = (lane >> 4) * 16;
#pragma unroll
      for (int kt = 0; kt < 8; ++kt) {
        if (m < EPB) {
          const _Float16* hp = &s_h1[m][kt * 32 + koff];
          at[kt].f4[0] = ((const float4*)hp)[0];
          at[kt].f4[1] = ((const float4*)(hp + 8))[0];
        } else {
#pragma unroll
          for (int q = 0; q < 16; ++q) at[kt].h[q] = (_Float16)0.f;
        }
      }
      const int NT = (phase == 0) ? (NP1 / 16) : (NP2 / 16);
      const _Float16* Wh = (phase == 0) ? Wp1 : Wp2;
      const float* bp = (phase == 0) ? s_bp1 : s_bp2;
      for (int nt = wv; nt < NT; nt += 8) {
        const int n0 = nt * 16;
        const _Float16* wrow =
            Wh + (size_t)(n0 + (lane & 15)) * HDIM + (lane >> 4) * 16;
        __builtin_prefetch(wrow + (size_t)128 * HDIM, 0, 1); // next tile
        H16 bt[8];
#pragma unroll
        for (int kt = 0; kt < 8; ++kt) {
          bt[kt].f4[0] = ((const float4*)(wrow + kt * 32))[0];
          bt[kt].f4[1] = ((const float4*)(wrow + kt * 32 + 8))[0];
        }
        v8f c;
#pragma unroll
        for (int r = 0; r < 8; ++r) c[r] = 0.f;
#pragma unroll
        for (int kt = 0; kt < 8; ++kt) c = wmma_f16(at[kt].v, bt[kt].v, c);
        if (lane < 16) {
          const int n = n0 + lane;
          const float bias = bp[n];
#pragma unroll
          for (int r = 0; r < 8; ++r) {
            float v = c[r] + bias;
            if (phase == 0) {
              if (n < 1024)
                s_bigA[r][n] =
                    alpha0 * v + (((n >> 5) == (n & 31)) ? 1.f : 0.f); // Am
              else if (n < 1280)
                s_Bm[r][n - 1024] = v;
              else
                s_nx[r][n - 1280] = MIN_VC + SPAN_VC / (1.f + __expf(-v));
            } else {
              if (n < 768)
                s_bigA[r][n] = v; // Cm
              else if (n < 792)
                s_na[r][n - 768] = MIN_VC + SPAN_VC / (1.f + __expf(-v));
            }
          }
        }
      }
    }
    __syncthreads();
  };

  // ---- prior step: one wave per element, lane = row, register-blocked.
  auto prior_valu = [&](int t) {
    float am[XDIM]; // own row of Am
#pragma unroll
    for (int j4 = 0; j4 < XDIM / 4; ++j4) {
      float4 v = ((const float4*)&s_bigA[e][i * XDIM])[j4];
      am[4 * j4 + 0] = v.x; am[4 * j4 + 1] = v.y;
      am[4 * j4 + 2] = v.z; am[4 * j4 + 3] = v.w;
    }
    { // prm = Am@qm + Bm@u[t-1]
      float s = 0.f;
#pragma unroll
      for (int j = 0; j < XDIM; ++j) s += am[j] * s_qm[e][j];
      const float* ut = u + ((size_t)(t - 1) * BATCH + bglob) * UDIM;
#pragma unroll
      for (int j = 0; j < UDIM; ++j) s += s_Bm[e][i * UDIM + j] * ut[j];
      s_prm[e][i] = s;
    }
    float tmp[XDIM]; // own row of Am@qc (never hits LDS)
#pragma unroll
    for (int k = 0; k < XDIM; ++k) tmp[k] = 0.f;
#pragma unroll
    for (int j = 0; j < XDIM; ++j) {
      const float aj = am[j];
      const float4* qr = (const float4*)&s_qc[e][j * XDIM]; // broadcast
#pragma unroll
      for (int k4 = 0; k4 < XDIM / 4; ++k4) {
        float4 q = qr[k4];
        tmp[4 * k4 + 0] += aj * q.x; tmp[4 * k4 + 1] += aj * q.y;
        tmp[4 * k4 + 2] += aj * q.z; tmp[4 * k4 + 3] += aj * q.w;
      }
    }
    float rawr[XDIM];
#pragma unroll
    for (int l = 0; l < XDIM; ++l) { // raw = Tmp @ Am^T + diag(nx)
      const float4* ar = (const float4*)&s_bigA[e][l * XDIM]; // broadcast
      float s = (i == l) ? s_nx[e][i] : 0.f;
#pragma unroll
      for (int k4 = 0; k4 < XDIM / 4; ++k4) {
        float4 av = ar[k4];
        s += tmp[4 * k4 + 0] * av.x + tmp[4 * k4 + 1] * av.y +
             tmp[4 * k4 + 2] * av.z + tmp[4 * k4 + 3] * av.w;
      }
      rawr[l] = s;
      s_raw[e][i * XDIM + l] = s;
    }
#pragma unroll
    for (int l = 0; l < XDIM; ++l) // prc = make_psd(raw)
      s_prc[e][i * XDIM + l] = 0.5f * (rawr[l] + s_raw[e][l * XDIM + i]) +
                               ((i == l) ? EPS_VC : 0.f);
  };

  // ---- posterior step small algebra + 24x24 SPD inverse
  auto post_valu = [&](int t) {
    if (i < ADIM) {
      float cm[XDIM]; // own row of Cm
#pragma unroll
      for (int j4 = 0; j4 < XDIM / 4; ++j4) {
        float4 v = ((const float4*)&s_bigA[e][i * XDIM])[j4];
        cm[4 * j4 + 0] = v.x; cm[4 * j4 + 1] = v.y;
        cm[4 * j4 + 2] = v.z; cm[4 * j4 + 3] = v.w;
      }
      float tmp[XDIM]; // own row of CmP = Cm@prc
#pragma unroll
      for (int k = 0; k < XDIM; ++k) tmp[k] = 0.f;
#pragma unroll
      for (int j = 0; j < XDIM; ++j) {
        const float cj = cm[j];
        const float4* pr = (const float4*)&s_prc[e][j * XDIM]; // broadcast
#pragma unroll
        for (int k4 = 0; k4 < XDIM / 4; ++k4) {
          float4 p = pr[k4];
          tmp[4 * k4 + 0] += cj * p.x; tmp[4 * k4 + 1] += cj * p.y;
          tmp[4 * k4 + 2] += cj * p.z; tmp[4 * k4 + 3] += cj * p.w;
        }
      }
#pragma unroll
      for (int k4 = 0; k4 < XDIM / 4; ++k4) { // CmP to LDS (needed x-lane)
        float4 o;
        o.x = tmp[4 * k4 + 0]; o.y = tmp[4 * k4 + 1];
        o.z = tmp[4 * k4 + 2]; o.w = tmp[4 * k4 + 3];
        ((float4*)&s_bigT[e][i * XDIM])[k4] = o;
      }
      { // innov = a_t - Cm @ prm
        float s = 0.f;
#pragma unroll
        for (int j = 0; j < XDIM; ++j) s += cm[j] * s_prm[e][j];
        s_iv[e][i] = a[((size_t)t * BATCH + bglob) * ADIM + i] - s;
      }
#pragma unroll
      for (int l = 0; l < ADIM; ++l) { // S = CmP @ Cm^T + na*I
        const float4* cr = (const float4*)&s_bigA[e][l * XDIM]; // broadcast
        float s2 = (i == l) ? s_na[e][i] : 0.f;
#pragma unroll
        for (int k4 = 0; k4 < XDIM / 4; ++k4) {
          float4 cv = cr[k4];
          s2 += tmp[4 * k4 + 0] * cv.x + tmp[4 * k4 + 1] * cv.y +
                tmp[4 * k4 + 2] * cv.z + tmp[4 * k4 + 3] * cv.w;
        }
        s_S[e][i * ADIM + l] = s2;
      }
    }
    // In-place Gauss-Jordan inverse (SPD, no pivoting); wave-lockstep safe.
    for (int k = 0; k < ADIM; ++k) {
      const float d = 1.f / s_S[e][k * ADIM + k];
      if (i == k) {
        for (int j = 0; j < ADIM; ++j)
          s_S[e][k * ADIM + j] = (j == k) ? d : s_S[e][k * ADIM + j] * d;
      }
      if (i < ADIM && i != k) {
        const float f = s_S[e][i * ADIM + k];
        for (int j = 0; j < ADIM; ++j)
          if (j != k) s_S[e][i * ADIM + j] -= f * s_S[e][k * ADIM + j];
        s_S[e][i * ADIM + k] = -f * d;
      }
    }
    // G row i (G = CmP^T @ Sinv) kept entirely in registers.
    float gr[ADIM];
#pragma unroll
    for (int l = 0; l < ADIM; ++l) gr[l] = 0.f;
#pragma unroll
    for (int k = 0; k < ADIM; ++k) {
      const float ck = s_bigT[e][k * XDIM + i];          // conflict-free col
      const float4* sr = (const float4*)&s_S[e][k * ADIM]; // broadcast row
#pragma unroll
      for (int l4 = 0; l4 < ADIM / 4; ++l4) {
        float4 sv = sr[l4];
        gr[4 * l4 + 0] += ck * sv.x; gr[4 * l4 + 1] += ck * sv.y;
        gr[4 * l4 + 2] += ck * sv.z; gr[4 * l4 + 3] += ck * sv.w;
      }
    }
    { // qm = prm + G @ innov
      float s = s_prm[e][i];
#pragma unroll
      for (int k = 0; k < ADIM; ++k) s += gr[k] * s_iv[e][k];
      s_qm[e][i] = s;
    }
    float acc[XDIM]; // raw = prc - G @ CmP (own row)
#pragma unroll
    for (int l4 = 0; l4 < XDIM / 4; ++l4) {
      float4 p = ((const float4*)&s_prc[e][i * XDIM])[l4];
      acc[4 * l4 + 0] = p.x; acc[4 * l4 + 1] = p.y;
      acc[4 * l4 + 2] = p.z; acc[4 * l4 + 3] = p.w;
    }
#pragma unroll
    for (int k = 0; k < ADIM; ++k) {
      const float gk = gr[k];
      const float4* tr = (const float4*)&s_bigT[e][k * XDIM]; // broadcast
#pragma unroll
      for (int l4 = 0; l4 < XDIM / 4; ++l4) {
        float4 tv = tr[l4];
        acc[4 * l4 + 0] -= gk * tv.x; acc[4 * l4 + 1] -= gk * tv.y;
        acc[4 * l4 + 2] -= gk * tv.z; acc[4 * l4 + 3] -= gk * tv.w;
      }
    }
#pragma unroll
    for (int l = 0; l < XDIM; ++l) s_raw[e][i * XDIM + l] = acc[l];
#pragma unroll
    for (int l = 0; l < XDIM; ++l) // qc = make_psd(raw)
      s_qc[e][i * XDIM + l] = 0.5f * (acc[l] + s_raw[e][l * XDIM + i]) +
                              ((i == l) ? EPS_VC : 0.f);
  };

  auto write_out = [&](int t) {
    __syncthreads();
    const size_t baseM = ((size_t)t * BATCH + (size_t)blockIdx.x * EPB) * XDIM;
    out[OPM + baseM + tid] = s_prm[tid >> 5][tid & 31];
    out[OQM + baseM + tid] = s_qm[tid >> 5][tid & 31];
    const size_t baseC =
        ((size_t)t * BATCH + (size_t)blockIdx.x * EPB) * (XDIM * XDIM);
    for (int r = 0; r < 32; ++r) {
      const int idx = r * 256 + tid;
      out[OPC + baseC + idx] = s_prc[idx >> 10][idx & 1023];
      out[OQC + baseC + idx] = s_qc[idx >> 10][idx & 1023];
    }
  };

  // ---- t = 0: prior = (0, I); posterior = post_step(0, I, a[0])
  s_prm[e][i] = 0.f;
  for (int l = 0; l < XDIM; ++l)
    s_prc[e][i * XDIM + l] = (i == l) ? 1.f : 0.f;
  dyn(&s_prm[0][0], 1);
  post_valu(0);
  write_out(0);

  for (int t = 1; t < TSTEPS; ++t) {
    dyn(&s_qm[0][0], 0);  // dyn(qm)
    prior_valu(t);        // uses u[t-1]
    dyn(&s_prm[0][0], 1); // dyn(prm)
    post_valu(t);         // uses a[t]
    write_out(t);
  }
}

// ---------------------------------------------------------------------------
extern "C" void kernel_launch(void* const* d_in, const int* in_sizes, int n_in,
                              void* d_out, int out_size, void* d_ws,
                              size_t ws_size, hipStream_t stream) {
  (void)in_sizes; (void)n_in; (void)out_size; (void)ws_size;
  const float* u   = (const float*)d_in[0];
  const float* a   = (const float*)d_in[1];
  const float* W1  = (const float*)d_in[2];
  const float* b1  = (const float*)d_in[3];
  const float* W2  = (const float*)d_in[4];
  const float* b2  = (const float*)d_in[5];
  const float* WA  = (const float*)d_in[6];
  const float* bA  = (const float*)d_in[7];
  const float* WB  = (const float*)d_in[8];
  const float* bB  = (const float*)d_in[9];
  const float* WC  = (const float*)d_in[10];
  const float* bC  = (const float*)d_in[11];
  const float* Wnx = (const float*)d_in[12];
  const float* bnx = (const float*)d_in[13];
  const float* Wna = (const float*)d_in[14];
  const float* bna = (const float*)d_in[15];
  const float* alpha = (const float*)d_in[16];

  char* ws = (char*)d_ws;
  _Float16* W1h = (_Float16*)(ws);          // 16384 B
  _Float16* W2h = (_Float16*)(ws + 16384);  // 131072 B
  _Float16* Wp1 = (_Float16*)(ws + 147456); // 671744 B
  _Float16* Wp2 = (_Float16*)(ws + 819200); // 409600 B
  float* bp1 = (float*)(ws + 1228800);      // 5248 B
  float* bp2 = (float*)(ws + 1234048);      // 3200 B

  pack_weights<<<128, 256, 0, stream>>>(W1, W2, WA, WB, WC, Wnx, Wna, bA, bB,
                                        bnx, bC, bna, W1h, W2h, Wp1, Wp2, bp1,
                                        bp2);
  kf_scan_kernel<<<NBLOCKS, 256, 0, stream>>>(u, a, b1, b2, bp1, bp2, alpha,
                                              W1h, W2h, Wp1, Wp2,
                                              (float*)d_out);
}
